// DeformableSpatialAttention_60189671686358
// MI455X (gfx1250) — matrix operand order repaired and checked by reading, not compiled, not run on hardware
//
#include <hip/hip_runtime.h>

// ---------------------------------------------------------------------------
// DeformableSpatialAttention for MI455X (gfx1250, wave32, WMMA)
//
// Pass 1: channel mean+max reduction of x (134 MB read)  -> att planes (1 MB)
// Pass 2: fused [7x7 conv (2->147 ch) as WMMA implicit GEMM] + deformable
//         conv + sigmoid, one wave per 16-pixel tile      -> att_final (0.5MB)
// Pass 3: out = x * att_final (134 MB read + 134 MB write, float4 streams)
//
// Roofline: ~0.4 GB HBM traffic total -> ~17us at 23.3 TB/s. Conv math is
// ~3.8 GFLOP, trivially hidden behind the streaming passes; done with
// v_wmma_f32_16x16x32_f16 (10 M-tiles x 4 K-steps per 16-pixel tile).
// ---------------------------------------------------------------------------

typedef __attribute__((ext_vector_type(16))) _Float16 v16h;
typedef __attribute__((ext_vector_type(8)))  _Float16 v8h;
typedef __attribute__((ext_vector_type(8)))  float    v8f;

#define BSZ   8
#define CSZ   256
#define HSZ   128
#define WSZ   128
#define HW    (HSZ * WSZ)          // 16384
#define BHW   (BSZ * HW)           // 131072
#define KKT   49                   // 7x7 taps
#define NOUT  147                  // 98 offset channels + 49 modulation
#define MPAD  160                  // 10 tiles of 16
#define KPAD  128                  // 4 WMMA K-steps of 32 (true K = 98)

// ---- workspace layout (bytes) ----
#define WS_ATT   0                          // 2*BHW floats (avg, max planes)
#define WS_WPAD  (2 * BHW * 4)              // 160*128 f16  (padded GEMM weights)
#define WS_ATTF  (WS_WPAD + 64 * 1024)      // BHW floats   (final attention)

// ---------------------------------------------------------------------------
// Pack offset_w (98,2,7,7) and mod_w (49,2,7,7) into padded f16 A-matrix
// Wpad[m][k]: m = output channel (147 used), k = c*49 + ki*7 + kj (98 used).
// ---------------------------------------------------------------------------
__global__ void pack_weights(const float* __restrict__ ow,
                             const float* __restrict__ mw,
                             _Float16* __restrict__ wpad) {
    for (int idx = threadIdx.x; idx < MPAD * KPAD; idx += blockDim.x) {
        int m = idx >> 7;
        int k = idx & 127;
        float v = 0.0f;
        if (k < 98 && m < NOUT)
            v = (m < 98) ? ow[m * 98 + k] : mw[(m - 98) * 98 + k];
        wpad[idx] = (_Float16)v;
    }
}

// ---------------------------------------------------------------------------
// Channel reduction: avg & max over C=256. One block per (b,y) row,
// 128 threads over W -> coalesced loads, stride HW across channels.
// ---------------------------------------------------------------------------
__global__ void reduce_kernel(const float* __restrict__ x,
                              float* __restrict__ att) {
    int row  = blockIdx.x;            // b*H + y
    int w    = threadIdx.x;           // 0..127
    long base = (long)(row >> 7) * CSZ * HW + (long)(row & (HSZ - 1)) * WSZ + w;
    float s = 0.0f, mx = -3.402823466e38f;
    #pragma unroll 4
    for (int c = 0; c < CSZ; ++c) {
        float v = x[base + (long)c * HW];
        s += v;
        mx = fmaxf(mx, v);
    }
    att[row * WSZ + w]       = s * (1.0f / (float)CSZ);  // avg plane
    att[BHW + row * WSZ + w] = mx;                        // max plane
}

__device__ __forceinline__ float sigmoidf(float z) {
    return 1.0f / (1.0f + __expf(-z));
}

// bilinear sample with zero outside (matches reference's valid-mask gather)
__device__ __forceinline__ float bilin(const float* __restrict__ img,
                                       float py, float px) {
    float y0f = floorf(py), x0f = floorf(px);
    int y0 = (int)y0f, x0 = (int)x0f;
    float wy1 = py - y0f, wy0 = 1.0f - wy1;
    float wx1 = px - x0f, wx0 = 1.0f - wx1;
    int y1 = y0 + 1, x1 = x0 + 1;
    float v00 = (y0 >= 0 && y0 < HSZ && x0 >= 0 && x0 < WSZ) ? img[y0 * WSZ + x0] : 0.0f;
    float v01 = (y0 >= 0 && y0 < HSZ && x1 >= 0 && x1 < WSZ) ? img[y0 * WSZ + x1] : 0.0f;
    float v10 = (y1 >= 0 && y1 < HSZ && x0 >= 0 && x0 < WSZ) ? img[y1 * WSZ + x0] : 0.0f;
    float v11 = (y1 >= 0 && y1 < HSZ && x1 >= 0 && x1 < WSZ) ? img[y1 * WSZ + x1] : 0.0f;
    return v00 * wy0 * wx0 + v01 * wy0 * wx1 + v10 * wy1 * wx0 + v11 * wy1 * wx1;
}

// ---------------------------------------------------------------------------
// Fused conv (WMMA) + deformable conv + sigmoid.
// One wave (32 threads) per 16-pixel tile along W.
//   GEMM: D[160][16] = Wpad[160][128] x Patch[128][16]
//   A lane layout (16-bit 16x32, per ISA 7.12.2): lane L: m=L%16, g=L/16,
//     halves h: K = (h<8 ? h : 8+h) + 8g  -> two contiguous b128 loads.
//   B lane layout (same K order as A, per ISA B-matrix pattern):
//     lane L: n=L%16, halves h: K = (h<8 ? h : 8+h) + 8*(L/16).
// Results staged through LDS, then 16 lanes do per-pixel deform sampling.
// ---------------------------------------------------------------------------
__global__ void __launch_bounds__(32)
conv_deform(const float* __restrict__ att,
            const _Float16* __restrict__ wpad,
            const float* __restrict__ offset_b,
            const float* __restrict__ mod_b,
            const float* __restrict__ dconv_w,
            float* __restrict__ att_final) {
    __shared__ float sout[MPAD * 16];          // 10 KB: 160 channels x 16 pixels

    int lane = threadIdx.x;                    // 0..31 (one full wave)
    int g    = lane >> 4;
    int nn   = lane & 15;

    int tix = blockIdx.x;                      // BHW/16 tiles
    int row = tix >> 3;                        // b*H + y   (8 tiles per row)
    int y   = row & (HSZ - 1);
    int b   = row >> 7;
    int xx  = ((tix & 7) << 4) + nn;           // this lane-column's pixel x

    const float* attb = att + b * HW;          // + c*BHW for channel plane

    // ---- build im2col B fragments (K padded 98 -> 128, 4 chunks of 32) ----
    v16h breg[4];
    #pragma unroll
    for (int s = 0; s < 4; ++s) {
        int ks = 32 * s + 8 * g;
        #pragma unroll
        for (int h = 0; h < 16; ++h) {
            int k = ks + (h < 8 ? h : h + 8);
            float v = 0.0f;
            if (k < 98) {
                int c  = (k >= 49) ? 1 : 0;
                int t  = k - c * 49;
                int ki = t / 7, kj = t - ki * 7;
                int yy = y + ki - 3, xs = xx + kj - 3;
                if (yy >= 0 && yy < HSZ && xs >= 0 && xs < WSZ)
                    v = attb[c * BHW + yy * WSZ + xs];
            }
            breg[s][h] = (_Float16)v;
        }
    }

    // ---- 10 M-tiles x 4 K-steps of v_wmma_f32_16x16x32_f16 ----
    for (int mt = 0; mt < 10; ++mt) {
        v8f acc = {};
        #pragma unroll
        for (int s = 0; s < 4; ++s) {
            const _Float16* wr = wpad + (mt * 16 + nn) * KPAD + 32 * s + 8 * g;
            v8h lo = *(const v8h*)(wr);        // K = 32s+8g   .. +7
            v8h hi = *(const v8h*)(wr + 16);   // K = 32s+16+8g.. +7
            v16h a;
            #pragma unroll
            for (int i = 0; i < 8; ++i) { a[i] = lo[i]; a[i + 8] = hi[i]; }
            acc = __builtin_amdgcn_wmma_f32_16x16x32_f16(
                /*neg_a=*/false, a, /*neg_b=*/false, breg[s],
                /*c_mod=*/(short)0, acc, /*reuse_a=*/false, /*reuse_b=*/false);
        }
        // D layout: lane L, VGPR v -> m = v + 8*(L/16), n = L%16
        #pragma unroll
        for (int v = 0; v < 8; ++v)
            sout[(mt * 16 + v + 8 * g) * 16 + nn] = acc[v];
    }
    __syncthreads();

    // ---- deformable conv + sigmoid: 16 lanes, one pixel each ----
    if (lane < 16) {
        float accd = 0.0f;
        for (int kk = 0; kk < KKT; ++kk) {
            float oy = sout[(2 * kk + 0) * 16 + nn] + offset_b[2 * kk + 0];
            float ox = sout[(2 * kk + 1) * 16 + nn] + offset_b[2 * kk + 1];
            float mv = 2.0f * sigmoidf(sout[(98 + kk) * 16 + nn] + mod_b[kk]);
            int ki = kk / 7, kj = kk - ki * 7;
            float py = (float)(y + ki - 3) + oy;
            float px = (float)(xx + kj - 3) + ox;
            float s0 = bilin(attb, py, px);          // avg channel
            float s1 = bilin(attb + BHW, py, px);    // max channel
            accd += mv * (dconv_w[kk] * s0 + dconv_w[49 + kk] * s1);
        }
        att_final[b * HW + y * WSZ + xx] = sigmoidf(accd);
    }
}

// ---------------------------------------------------------------------------
// out = x * att_final, float4-vectorized (the bandwidth-bound pass).
// ---------------------------------------------------------------------------
__global__ void apply_kernel(const float* __restrict__ x,
                             const float* __restrict__ attf,
                             float* __restrict__ out) {
    int i4 = blockIdx.x * blockDim.x + threadIdx.x;  // 0 .. B*C*HW/4 - 1
    int b  = i4 >> 20;                               // / (C*HW/4)
    int p4 = i4 & 4095;                              // (HW/4) - 1
    float4 xv = ((const float4*)x)[i4];
    float4 av = ((const float4*)attf)[b * 4096 + p4];
    float4 o;
    o.x = xv.x * av.x; o.y = xv.y * av.y;
    o.z = xv.z * av.z; o.w = xv.w * av.w;
    ((float4*)out)[i4] = o;
}

extern "C" void kernel_launch(void* const* d_in, const int* in_sizes, int n_in,
                              void* d_out, int out_size, void* d_ws, size_t ws_size,
                              hipStream_t stream) {
    (void)in_sizes; (void)n_in; (void)out_size; (void)ws_size;
    const float* x        = (const float*)d_in[0];
    const float* offset_w = (const float*)d_in[1];
    const float* offset_b = (const float*)d_in[2];
    const float* mod_w    = (const float*)d_in[3];
    const float* mod_b    = (const float*)d_in[4];
    const float* dconv_w  = (const float*)d_in[5];
    float* out = (float*)d_out;

    char* ws        = (char*)d_ws;
    float*    att   = (float*)(ws + WS_ATT);      // 2*BHW floats (1 MB)
    _Float16* wpad  = (_Float16*)(ws + WS_WPAD);  // 160*128 f16 (40 KB)
    float*    attf  = (float*)(ws + WS_ATTF);     // BHW floats (0.5 MB)

    pack_weights<<<1, 256, 0, stream>>>(offset_w, mod_w, wpad);
    reduce_kernel<<<BSZ * HSZ, WSZ, 0, stream>>>(x, att);
    conv_deform<<<BHW / 16, 32, 0, stream>>>(att, wpad, offset_b, mod_b,
                                             dconv_w, attf);
    apply_kernel<<<(BSZ * CSZ * HW / 4) / 256, 256, 0, stream>>>(x, attf, out);
}
